// RNN_326417515127
// MI455X (gfx1250) — matrix operand order repaired and checked by reading, not compiled, not run on hardware
//
#include <hip/hip_runtime.h>
#include <hip/hip_bf16.h>

#define INPUT_SZ  1024
#define HIDDEN_SZ 1024
#define BATCH     64
#define SEQ       512
#define WROW      2048          // W row length (INPUT_SZ + HIDDEN_SZ)

typedef __bf16  v16bf __attribute__((ext_vector_type(16)));
typedef __bf16  v8bf  __attribute__((ext_vector_type(8)));   // 16-byte chunk
typedef float   v8f   __attribute__((ext_vector_type(8)));
typedef unsigned int u32x4 __attribute__((ext_vector_type(4)));
typedef int          i32x4 __attribute__((ext_vector_type(4)));
typedef int          i32x8 __attribute__((ext_vector_type(8)));

// ---------------------------------------------------------------------------
// LDS generic pointer -> 32-bit LDS byte address (shared aperture: addr[31:0])
__device__ inline unsigned lds_addr_of(const void* p) {
    return (unsigned)(uintptr_t)p;
}

// CDNA5 async copy: per-lane 16B global -> LDS, tracked by ASYNCcnt.
__device__ inline void async_b128(unsigned lds_byte, const void* gsrc) {
    asm volatile("global_load_async_to_lds_b128 %0, %1, off"
                 :: "v"(lds_byte), "v"(gsrc) : "memory");
}
__device__ inline void wait_async0() {
    asm volatile("s_wait_asynccnt 0x0" ::: "memory");
}
__device__ inline void wait_async4() {
    asm volatile("s_wait_asynccnt 0x4" ::: "memory");
}

// ---------------------------------------------------------------------------
// WMMA fragment loaders (bf16 16x16x32, wave32), per ISA 7.12.2 layouts.
__device__ inline v16bf load_a_frag(const __bf16* base, int ldk, int lane) {
    int m  = lane & 15;
    int kh = lane >> 4;
    const __bf16* p = base + (size_t)m * ldk + kh * 8;
    v8bf lo = *(const v8bf*)(p);
    v8bf hi = *(const v8bf*)(p + 16);
    v16bf r;
    #pragma unroll
    for (int i = 0; i < 8; ++i) { r[i] = lo[i]; r[i + 8] = hi[i]; }
    return r;
}

__device__ inline v16bf load_b_frag(const __bf16* base, int ldk, int lane) {
    int n  = lane & 15;
    int kh = lane >> 4;
    const __bf16* p = base + (size_t)n * ldk + kh * 8;
    v8bf lo = *(const v8bf*)(p);
    v8bf hi = *(const v8bf*)(p + 16);
    v16bf r;
    #pragma unroll
    for (int i = 0; i < 8; ++i) { r[i] = lo[i]; r[i + 8] = hi[i]; }
    return r;
}

__device__ inline v8f wmma_bf16(v16bf a, v16bf b, v8f c) {
    return __builtin_amdgcn_wmma_f32_16x16x32_bf16(
        false, a, false, b, (short)0, c, false, false);
}

// ---------------------------------------------------------------------------
__global__ void f32_to_bf16(const float* __restrict__ src,
                            __bf16* __restrict__ dst, size_t n) {
    size_t i = (size_t)blockIdx.x * blockDim.x + threadIdx.x;
    size_t stride = (size_t)gridDim.x * blockDim.x;
    for (; i < n; i += stride) dst[i] = (__bf16)src[i];
}

__global__ void init_ws(__bf16* h0, __bf16* h1, unsigned* cnt) {
    size_t i = (size_t)blockIdx.x * blockDim.x + threadIdx.x;
    if (i == 0) *cnt = 0u;
    size_t stride = (size_t)gridDim.x * blockDim.x;
    const size_t n = (size_t)BATCH * HIDDEN_SZ;
    for (size_t j = i; j < n; j += stride) {
        h0[j] = (__bf16)0.0f;
        h1[j] = (__bf16)0.0f;
    }
}

// ---------------------------------------------------------------------------
// Phase 1: xp = x @ Wx^T + b -> fp32 into d_out (reused as xp storage).
// 128x128 tile/WG, double-buffered LDS, async-to-LDS staging pipelined
// against the WMMA K loop (4 async instrs/wave/stage -> s_wait_asynccnt 4).
__global__ void __launch_bounds__(256)
gemm_xp(const __bf16* __restrict__ xb, const __bf16* __restrict__ Wbf,
        const float* __restrict__ bias, float* __restrict__ out) {
    __shared__ __bf16 lA[2][128 * 32];   // [m][k]  2 x 8 KB
    __shared__ __bf16 lB[2][128 * 32];   // [n][k]  2 x 8 KB

    const int r0 = blockIdx.x * 128;
    const int c0 = blockIdx.y * 128;
    const int tid  = threadIdx.x;
    const int wid  = tid >> 5;
    const int lane = tid & 31;
    const int wm = wid >> 1;             // 0..3
    const int wn = wid & 1;              // 0..1

    const unsigned baseA[2] = { lds_addr_of(&lA[0][0]), lds_addr_of(&lA[1][0]) };
    const unsigned baseB[2] = { lds_addr_of(&lB[0][0]), lds_addr_of(&lB[1][0]) };

    auto stage = [&](int buf, int kk) {
        #pragma unroll
        for (int i = tid; i < 512; i += 256) {           // A: 128 rows x 32 k
            int r = i >> 2, c8 = i & 3;
            async_b128(baseA[buf] + (unsigned)i * 16,
                       xb + (size_t)(r0 + r) * INPUT_SZ + kk + c8 * 8);
        }
        #pragma unroll
        for (int i = tid; i < 512; i += 256) {           // B: Wx rows are [N][K]
            int r = i >> 2, c8 = i & 3;
            async_b128(baseB[buf] + (unsigned)i * 16,
                       Wbf + (size_t)(c0 + r) * WROW + kk + c8 * 8);
        }
    };

    v8f acc[2][4] = {};
    stage(0, 0);
    for (int ki = 0; ki < 32; ++ki) {
        const int cur = ki & 1;
        if (ki + 1 < 32) { stage(cur ^ 1, (ki + 1) * 32); wait_async4(); }
        else             { wait_async0(); }
        __syncthreads();

        v16bf afr[2], bfr[4];
        #pragma unroll
        for (int im = 0; im < 2; ++im)
            afr[im] = load_a_frag(&lA[cur][(size_t)(wm * 32 + im * 16) * 32], 32, lane);
        #pragma unroll
        for (int in = 0; in < 4; ++in)
            bfr[in] = load_b_frag(&lB[cur][(size_t)(wn * 64 + in * 16) * 32], 32, lane);
        #pragma unroll
        for (int im = 0; im < 2; ++im)
            #pragma unroll
            for (int in = 0; in < 4; ++in)
                acc[im][in] = wmma_bf16(afr[im], bfr[in], acc[im][in]);
        __syncthreads();
    }

    const int n = lane & 15;
    const int mbase = (lane < 16) ? 0 : 8;
    #pragma unroll
    for (int im = 0; im < 2; ++im) {
        #pragma unroll
        for (int in = 0; in < 4; ++in) {
            int gcol = c0 + wn * 64 + in * 16 + n;
            float bv = bias[gcol];
            #pragma unroll
            for (int v = 0; v < 8; ++v) {
                int grow = r0 + wm * 32 + im * 16 + mbase + v;
                out[(size_t)grow * HIDDEN_SZ + gcol] = acc[im][in][v] + bv;
            }
        }
    }
}

// ---------------------------------------------------------------------------
__device__ inline void grid_barrier(unsigned* cnt, unsigned nwg, unsigned step) {
    __syncthreads();
    if (threadIdx.x == 0) {
        __threadfence();
        atomicAdd(cnt, 1u);
        unsigned target = (step + 1u) * nwg;
        while (atomicAdd(cnt, 0u) < target) { __builtin_amdgcn_s_sleep(1); }
        __threadfence();
    }
    __syncthreads();
}

// ---------------------------------------------------------------------------
// Phase 2: persistent scan. 64 WGs = 4 batch-groups(16) x 16 col-groups(64).
// Wh slice (64x1024 bf16 = 128 KB) loaded ONCE into LDS via the Tensor Data
// Mover (2-D D#: tile 1024x64, dim0_stride 2048), resident for all 512 steps.
__global__ void __launch_bounds__(128)
rnn_scan(const __bf16* __restrict__ Wbf, float* __restrict__ out,
         __bf16* __restrict__ h0, __bf16* __restrict__ h1,
         unsigned* __restrict__ cnt) {
    __shared__ __bf16 lWh[64 * 1024];  // [n][k] slice of Wh, 128 KB
    __shared__ __bf16 lH[16 * 1024];   // h tile for this batch group, 32 KB

    const int wg = blockIdx.x;         // 0..63
    const int bg = wg >> 4;            // batch group 0..3
    const int cg = wg & 15;            // column group 0..15
    const int tid  = threadIdx.x;
    const int wid  = tid >> 5;         // 0..3 (one 16-col N-tile per wave)
    const int lane = tid & 31;

#if __has_builtin(__builtin_amdgcn_tensor_load_to_lds)
    if (wid == 0) {
        unsigned long long ga = (unsigned long long)(uintptr_t)
            (Wbf + (size_t)(cg * 64) * WROW + INPUT_SZ);
        unsigned lo = (unsigned)__builtin_amdgcn_readfirstlane((int)(ga & 0xffffffffu));
        unsigned hi = (unsigned)__builtin_amdgcn_readfirstlane((int)(ga >> 32));
        u32x4 g0 = {};
        g0[0] = 1u;                                 // count=1, user D#
        g0[1] = lds_addr_of(lWh);                   // lds_addr
        g0[2] = lo;                                 // global_addr[31:0]
        g0[3] = (hi & 0x01ffffffu) | (2u << 30);    // global_addr[56:32] | type=2
        i32x8 g1 = {};
        g1[0] = 1 << 16;                            // data_size = 2 bytes
        g1[1] = 1024 << 16;                         // tensor_dim0 = 1024
        g1[2] = 64 << 16;                           // tensor_dim1 = 64
        g1[3] = 1024 << 16;                         // tile_dim0 = 1024
        g1[4] = 64;                                 // tile_dim1 = 64
        g1[5] = WROW;                               // tensor_dim0_stride = 2048
        i32x4 g2 = {}, g3 = {};
#if __clang_major__ >= 23
        i32x8 g4 = {};
        __builtin_amdgcn_tensor_load_to_lds(g0, g1, g2, g3, g4, 0);
#else
        __builtin_amdgcn_tensor_load_to_lds(g0, g1, g2, g3, 0);
#endif
        __builtin_amdgcn_s_wait_tensorcnt(0);
    }
#else
    for (int i = tid; i < 64 * 1024 / 8; i += 128) {
        int r = i >> 7, c8 = i & 127;
        async_b128(lds_addr_of(lWh) + (unsigned)i * 16,
                   Wbf + (size_t)(cg * 64 + r) * WROW + INPUT_SZ + c8 * 8);
    }
#endif

    const int n = lane & 15;
    const int mbase = (lane < 16) ? 0 : 8;
    const int gcol = cg * 64 + wid * 16 + n;
    const unsigned lH_base = lds_addr_of(lH);

    for (int t = 0; t < SEQ; ++t) {
        const __bf16* hc = (t & 1) ? h1 : h0;
        __bf16*       hn = (t & 1) ? h0 : h1;

        // async-stage h tile (16 x 1024 bf16 = 32 KB; 16 instrs/wave)
        for (int i = tid; i < 16 * 1024 / 8; i += 128) {
            int r = i >> 7, c8 = i & 127;
            async_b128(lH_base + (unsigned)i * 16,
                       hc + (size_t)(bg * 16 + r) * HIDDEN_SZ + c8 * 8);
        }
        wait_async0();
        __syncthreads();

        // get the fp32 xp row moving while the K loop runs
        __builtin_prefetch(
            &out[((size_t)(bg * 16 + mbase) * SEQ + t) * HIDDEN_SZ + gcol], 0, 0);

        v8f acc = {};
        #pragma unroll 4
        for (int k0 = 0; k0 < HIDDEN_SZ; k0 += 32) {
            v16bf a = load_a_frag(lH + k0, HIDDEN_SZ, lane);
            v16bf b = load_b_frag(lWh + (size_t)(wid * 16) * HIDDEN_SZ + k0,
                                  HIDDEN_SZ, lane);
            acc = wmma_bf16(a, b, acc);
        }

        #pragma unroll
        for (int v = 0; v < 8; ++v) {
            int b = bg * 16 + mbase + v;
            size_t oidx = ((size_t)b * SEQ + t) * HIDDEN_SZ + gcol;
            float hval = tanhf(out[oidx] + acc[v]);
            out[oidx] = hval;
            hn[(size_t)b * HIDDEN_SZ + gcol] = (__bf16)hval;
        }

        grid_barrier(cnt, 64u, (unsigned)t);
    }
}

// ---------------------------------------------------------------------------
extern "C" void kernel_launch(void* const* d_in, const int* in_sizes, int n_in,
                              void* d_out, int out_size, void* d_ws, size_t ws_size,
                              hipStream_t stream) {
    const float* x = (const float*)d_in[0];   // (64,512,1024)
    const float* W = (const float*)d_in[1];   // (1024,2048)
    const float* b = (const float*)d_in[2];   // (1024,)
    float* out = (float*)d_out;               // (64,512,1024), reused as xp

    char* ws = (char*)d_ws;
    __bf16* Wbf = (__bf16*)ws;                                    // 4 MB
    __bf16* xb  = (__bf16*)(ws + (size_t)4 * 1024 * 1024);        // 64 MB
    __bf16* h0  = (__bf16*)(ws + (size_t)68 * 1024 * 1024);       // 128 KB
    __bf16* h1  = h0 + (size_t)BATCH * HIDDEN_SZ;                 // 128 KB
    unsigned* cnt = (unsigned*)(h1 + (size_t)BATCH * HIDDEN_SZ);

    f32_to_bf16<<<2048, 256, 0, stream>>>(W, Wbf, (size_t)HIDDEN_SZ * WROW);
    f32_to_bf16<<<8192, 256, 0, stream>>>(x, xb,
                                          (size_t)BATCH * SEQ * INPUT_SZ);
    init_ws<<<64, 256, 0, stream>>>(h0, h1, cnt);

    dim3 g1(BATCH * SEQ / 128, HIDDEN_SZ / 128);   // 256 x 8
    gemm_xp<<<g1, 256, 0, stream>>>(xb, Wbf, b, out);

    rnn_scan<<<64, 128, 0, stream>>>(Wbf, out, h0, h1, cnt);
}
